// LinearAttention_13984413516285
// MI455X (gfx1250) — compile-verified, matrix operand order checked by reading, not run on hardware
//
#include <hip/hip_runtime.h>

#define DD 512
#define TT 4096
#define BB 2
#define HH 8
#define DH 64
#define NROW (BB * TT)   // 8192
#define CH 64            // chunk length

typedef __attribute__((ext_vector_type(16))) __bf16 v16bf;
typedef __attribute__((ext_vector_type(8)))  float  v8f;
typedef __attribute__((ext_vector_type(8)))  unsigned short v8u;
typedef __attribute__((ext_vector_type(16))) unsigned short v16u;

__device__ __forceinline__ unsigned short f2bf(float f) {
  unsigned u = __builtin_bit_cast(unsigned, f);
  u += 0x7FFFu + ((u >> 16) & 1u);           // round-to-nearest-even
  return (unsigned short)(u >> 16);
}
__device__ __forceinline__ float bf2f(unsigned short s) {
  unsigned u = ((unsigned)s) << 16;
  return __builtin_bit_cast(float, u);
}

// 16x32 bf16 A-fragment (or B-fragment from B^T) from row-major u16 storage.
// Lane L (<16): row L, K ko..ko+7 -> elems 0..7, K ko+16..ko+23 -> elems 8..15
// Lane L+16: same row, ko = 8. (CDNA5 ISA 7.12.2, 16-bit A-matrix 16x32)
__device__ __forceinline__ v16bf load_frag_u16(const unsigned short* base, int ld) {
  const int lane = threadIdx.x & 31;
  const int r  = lane & 15;
  const int ko = (lane >> 4) << 3;
  const unsigned short* p = base + r * ld + ko;
  v8u lo = *(const v8u*)(p);
  v8u hi = *(const v8u*)(p + 16);
  v16u t;
#pragma unroll
  for (int i = 0; i < 8; ++i) { t[i] = lo[i]; t[i + 8] = hi[i]; }
  return __builtin_bit_cast(v16bf, t);
}

// Same fragment, sourced from f32 (running state in LDS), converted to bf16.
__device__ __forceinline__ v16bf load_frag_f32(const float* base, int ld) {
  const int lane = threadIdx.x & 31;
  const int r  = lane & 15;
  const int ko = (lane >> 4) << 3;
  const float* p = base + r * ld + ko;
  v16u t;
#pragma unroll
  for (int i = 0; i < 8; ++i) { t[i] = f2bf(p[i]); t[i + 8] = f2bf(p[16 + i]); }
  return __builtin_bit_cast(v16bf, t);
}

#define WMMA_BF16(a, b, c) \
  __builtin_amdgcn_wmma_f32_16x16x32_bf16(false, (a), false, (b), (short)0, (c), false, false)

// ---- CDNA5 async global->LDS copy (ASYNCcnt-tracked), per ISA 10.x / 15.18 ----
__device__ __forceinline__ void async_copy_b128(const unsigned short* lds_dst,
                                                const unsigned short* gsrc) {
  const unsigned ldsoff = (unsigned)(uintptr_t)lds_dst;     // low 32 bits = LDS offset
  const unsigned long long ga = (unsigned long long)(uintptr_t)gsrc;
  asm volatile("global_load_async_to_lds_b128 %0, %1, off"
               :: "v"(ldsoff), "v"(ga) : "memory");
}
__device__ __forceinline__ void wait_async0() {
  asm volatile("s_wait_asynccnt 0x0" ::: "memory");
}

// ---------------------------------------------------------------- convert
__global__ __launch_bounds__(256) void cvt_bf16_kernel(const float* __restrict__ in,
                                                       unsigned short* __restrict__ out, int n) {
  int i = blockIdx.x * 256 + threadIdx.x;
  if (i < n) out[i] = f2bf(in[i]);
}

// ------------------------------------------------- fused QKV projection + elu+1
// out[n,d] = x[n,:] . W[d,:]  (x @ W^T): B^T rows are W rows -> contiguous.
__global__ __launch_bounds__(256) void proj_qkv_kernel(
    const unsigned short* __restrict__ xbf,
    const unsigned short* __restrict__ wq, const unsigned short* __restrict__ wk,
    const unsigned short* __restrict__ wv,
    const float* __restrict__ bq, const float* __restrict__ bk, const float* __restrict__ bv,
    unsigned short* __restrict__ qb, unsigned short* __restrict__ kb,
    unsigned short* __restrict__ vb) {
  const int wave = blockIdx.x * 8 + (threadIdx.x >> 5);  // 16384 tiles: 512 row x 32 col
  const int rt = wave >> 5, ct = wave & 31;
  const int row0 = rt << 4, col0 = ct << 4;
  v8f cq = {}, ck = {}, cv = {};
  for (int kk = 0; kk < DD; kk += 32) {
    if (kk + 32 < DD) __builtin_prefetch(xbf + row0 * DD + kk + 32, 0, 3);
    v16bf a  = load_frag_u16(xbf + row0 * DD + kk, DD);
    v16bf fq = load_frag_u16(wq  + col0 * DD + kk, DD);
    v16bf fk = load_frag_u16(wk  + col0 * DD + kk, DD);
    v16bf fv = load_frag_u16(wv  + col0 * DD + kk, DD);
    cq = WMMA_BF16(a, fq, cq);
    ck = WMMA_BF16(a, fk, ck);
    cv = WMMA_BF16(a, fv, cv);
  }
  const int lane = threadIdx.x & 31;
  const int n = lane & 15, mb = (lane >> 4) << 3;   // C-frag: elem i -> (mb+i, n)
  const int col = col0 + n;
  const float biq = bq[col], bik = bk[col], biv = bv[col];
#pragma unroll
  for (int i = 0; i < 8; ++i) {
    const int row = row0 + mb + i;
    float q = cq[i] + biq;  q = (q > 0.f) ? (q + 1.f) : __expf(q);  // elu(q)+1
    float k = ck[i] + bik;  k = (k > 0.f) ? (k + 1.f) : __expf(k);
    qb[row * DD + col] = f2bf(q);
    kb[row * DD + col] = f2bf(k);
    vb[row * DD + col] = f2bf(cv[i] + biv);
  }
}

// ------------------------------------------------- chunked causal linear attention
// One block per (b,h). State S kept transposed in LDS: ST[j][i] = S[i][j] (f32).
// Q/K/V chunks staged into LDS via CDNA5 async global->LDS copies.
__global__ __launch_bounds__(256) void linattn_kernel(
    const unsigned short* __restrict__ qb, const unsigned short* __restrict__ kb,
    const unsigned short* __restrict__ vb, unsigned short* __restrict__ ob) {
  __shared__ __align__(16) float  stateT[DH * DH];     // 16 KB
  __shared__ __align__(16) float  zvec[DH];
  __shared__ __align__(16) float  invden[CH];
  __shared__ __align__(16) unsigned short QR[CH * DH]; // Q chunk, row-major [t][d]
  __shared__ __align__(16) unsigned short KR[CH * DH]; // K chunk, row-major [t][d]
  __shared__ __align__(16) unsigned short VR[CH * DH]; // V chunk, row-major [t][d]
  __shared__ __align__(16) unsigned short KT[DH * CH]; // K^T: [d][t]
  __shared__ __align__(16) unsigned short VT[DH * CH]; // V^T: [d][t]
  __shared__ __align__(16) unsigned short Pm[CH * CH]; // masked P, bf16

  const int bh = blockIdx.x;            // 0..15
  const int b = bh >> 3, h = bh & 7;
  const int tid = threadIdx.x;          // 0..255 (8 waves)
  const int wid = tid >> 5, lane = tid & 31;
  const int n = lane & 15, mb = (lane >> 4) << 3;

  for (int i = tid; i < DH * DH; i += 256) stateT[i] = 0.f;
  if (tid < DH) zvec[tid] = 0.f;
  __syncthreads();

  const size_t headoff = (size_t)b * TT * DD + (size_t)h * DH;
  for (int c0 = 0; c0 < TT; c0 += CH) {
    const unsigned short* Qc = qb + headoff + (size_t)c0 * DD;
    const unsigned short* Kc = kb + headoff + (size_t)c0 * DD;
    const unsigned short* Vc = vb + headoff + (size_t)c0 * DD;

    // async-stage Q,K,V chunks (row-major) into LDS: 512 x 16B transfers each
    for (int j = tid; j < CH * 8; j += 256) {
      const int t = j >> 3, seg = (j & 7) << 3;       // 8 bf16 per transfer
      async_copy_b128(&QR[t * CH + seg], Qc + (size_t)t * DD + seg);
      async_copy_b128(&KR[t * CH + seg], Kc + (size_t)t * DD + seg);
      async_copy_b128(&VR[t * CH + seg], Vc + (size_t)t * DD + seg);
    }
    wait_async0();
    __syncthreads();

    // build K^T, V^T in LDS
    for (int i = tid; i < CH * DH; i += 256) {
      int t = i >> 6, d = i & 63;
      KT[d * CH + t] = KR[t * CH + d];
      VT[d * CH + t] = VR[t * CH + d];
    }
    __syncthreads();

    // P = Q K^T, causal mask, store bf16 to Pm. 16 tiles, 2 per wave.
    for (int tile = wid; tile < 16; tile += 8) {
      const int tm = tile >> 2, tn = tile & 3;        // rows t, cols s
      v8f p = {};
      p = WMMA_BF16(load_frag_u16(QR + tm * 16 * CH, CH),
                    load_frag_u16(KR + tn * 16 * CH, CH), p);
      p = WMMA_BF16(load_frag_u16(QR + tm * 16 * CH + 32, CH),
                    load_frag_u16(KR + tn * 16 * CH + 32, CH), p);
#pragma unroll
      for (int i = 0; i < 8; ++i) {
        const int tr = tm * 16 + mb + i, sc = tn * 16 + n;
        Pm[tr * CH + sc] = f2bf((sc <= tr) ? p[i] : 0.f);
      }
    }
    __syncthreads();

    // den_t = q_t . z_prev + rowsum(Pm)   (threads 0..63)
    if (tid < CH) {
      float rs = 0.f;
      for (int s = 0; s < CH; ++s) rs += bf2f(Pm[tid * CH + s]);
      float dz = 0.f;
      for (int d = 0; d < DH; ++d) dz += bf2f(QR[tid * CH + d]) * zvec[d];
      float den = dz + rs;
      invden[tid] = 1.f / fmaxf(den, 1e-6f);
    }
    __syncthreads();

    // O = (Q @ S_prev + Pm @ V) * invden
    for (int tile = wid; tile < 16; tile += 8) {
      const int tm = tile >> 2, tn = tile & 3;        // rows t, cols d_v
      v8f acc = {};
      // inter: A = Q (K=i), B^T = stateT rows j (K=i)
      acc = WMMA_BF16(load_frag_u16(QR + tm * 16 * CH, CH),
                      load_frag_f32(stateT + tn * 16 * DH, DH), acc);
      acc = WMMA_BF16(load_frag_u16(QR + tm * 16 * CH + 32, CH),
                      load_frag_f32(stateT + tn * 16 * DH + 32, DH), acc);
      // intra: A = Pm (K=s), B^T = V^T rows d (K=s)
      acc = WMMA_BF16(load_frag_u16(Pm + tm * 16 * CH, CH),
                      load_frag_u16(VT + tn * 16 * CH, CH), acc);
      acc = WMMA_BF16(load_frag_u16(Pm + tm * 16 * CH + 32, CH),
                      load_frag_u16(VT + tn * 16 * CH + 32, CH), acc);
#pragma unroll
      for (int i = 0; i < 8; ++i) {
        const int t = tm * 16 + mb + i;
        ob[headoff + (size_t)(c0 + t) * DD + tn * 16 + n] = f2bf(acc[i] * invden[t]);
      }
    }
    __syncthreads();

    // state update: ST[j][i] += sum_t V^T[j][t] * K^T[i][t]   (C-operand accumulate)
    for (int tile = wid; tile < 16; tile += 8) {
      const int tj = tile >> 2, ti = tile & 3;
      v8f cf;
#pragma unroll
      for (int i = 0; i < 8; ++i) cf[i] = stateT[(tj * 16 + mb + i) * DH + ti * 16 + n];
      cf = WMMA_BF16(load_frag_u16(VT + tj * 16 * CH, CH),
                     load_frag_u16(KT + ti * 16 * CH, CH), cf);
      cf = WMMA_BF16(load_frag_u16(VT + tj * 16 * CH + 32, CH),
                     load_frag_u16(KT + ti * 16 * CH + 32, CH), cf);
#pragma unroll
      for (int i = 0; i < 8; ++i) stateT[(tj * 16 + mb + i) * DH + ti * 16 + n] = cf[i];
    }
    if (tid < DH) {
      float za = 0.f;
      for (int t = 0; t < CH; ++t) za += bf2f(KT[tid * CH + t]);
      zvec[tid] += za;
    }
    __syncthreads();
  }
}

// ------------------------------------------------- output projection (f32 out)
__global__ __launch_bounds__(256) void proj_out_kernel(
    const unsigned short* __restrict__ ab, const unsigned short* __restrict__ wo,
    const float* __restrict__ bo, float* __restrict__ out) {
  const int wave = blockIdx.x * 8 + (threadIdx.x >> 5);
  const int rt = wave >> 5, ct = wave & 31;
  const int row0 = rt << 4, col0 = ct << 4;
  v8f c = {};
  for (int kk = 0; kk < DD; kk += 32) {
    if (kk + 32 < DD) __builtin_prefetch(ab + row0 * DD + kk + 32, 0, 3);
    v16bf a  = load_frag_u16(ab + row0 * DD + kk, DD);
    v16bf bf = load_frag_u16(wo + col0 * DD + kk, DD);
    c = WMMA_BF16(a, bf, c);
  }
  const int lane = threadIdx.x & 31;
  const int n = lane & 15, mb = (lane >> 4) << 3;
  const int col = col0 + n;
  const float bias = bo[col];
#pragma unroll
  for (int i = 0; i < 8; ++i)
    out[(row0 + mb + i) * DD + col] = c[i] + bias;
}

// ----------------------------------------------------------------------------
extern "C" void kernel_launch(void* const* d_in, const int* in_sizes, int n_in,
                              void* d_out, int out_size, void* d_ws, size_t ws_size,
                              hipStream_t stream) {
  (void)in_sizes; (void)n_in; (void)out_size; (void)ws_size;
  const float* x  = (const float*)d_in[0];
  const float* Wq = (const float*)d_in[1];
  const float* bq = (const float*)d_in[2];
  const float* Wk = (const float*)d_in[3];
  const float* bk = (const float*)d_in[4];
  const float* Wv = (const float*)d_in[5];
  const float* bv = (const float*)d_in[6];
  const float* Wo = (const float*)d_in[7];
  const float* bo = (const float*)d_in[8];
  float* out = (float*)d_out;

  unsigned short* ws  = (unsigned short*)d_ws;
  unsigned short* xbf = ws;                        // 8192*512
  unsigned short* wqb = xbf + (size_t)NROW * DD;   // 512*512 each
  unsigned short* wkb = wqb + (size_t)DD * DD;
  unsigned short* wvb = wkb + (size_t)DD * DD;
  unsigned short* wob = wvb + (size_t)DD * DD;
  unsigned short* qb  = wob + (size_t)DD * DD;     // 8192*512 each
  unsigned short* kb  = qb  + (size_t)NROW * DD;
  unsigned short* vb  = kb  + (size_t)NROW * DD;
  unsigned short* ob  = vb  + (size_t)NROW * DD;

  const int NX = NROW * DD, NW = DD * DD;
  cvt_bf16_kernel<<<(NX + 255) / 256, 256, 0, stream>>>(x,  xbf, NX);
  cvt_bf16_kernel<<<(NW + 255) / 256, 256, 0, stream>>>(Wq, wqb, NW);
  cvt_bf16_kernel<<<(NW + 255) / 256, 256, 0, stream>>>(Wk, wkb, NW);
  cvt_bf16_kernel<<<(NW + 255) / 256, 256, 0, stream>>>(Wv, wvb, NW);
  cvt_bf16_kernel<<<(NW + 255) / 256, 256, 0, stream>>>(Wo, wob, NW);

  // 8192/16 = 512 row tiles, 512/16 = 32 col tiles -> 16384 wave-tiles / 8 waves
  proj_qkv_kernel<<<2048, 256, 0, stream>>>(xbf, wqb, wkb, wvb, bq, bk, bv, qb, kb, vb);
  linattn_kernel<<<BB * HH, 256, 0, stream>>>(qb, kb, vb, ob);
  proj_out_kernel<<<2048, 256, 0, stream>>>(ob, wob, bo, out);
}
